// MHF_78013785964705
// MI455X (gfx1250) — matrix operand order repaired and checked by reading, not compiled, not run on hardware
//
#include <hip/hip_runtime.h>
#include <hip/hip_bf16.h>

// ---------------------------------------------------------------------------
// MI455X / gfx1250: wave32, WMMA 16x16x32 bf16 (f32 accum), async-to-LDS
// double-buffered staging (ASYNCcnt), fused silu(Q K^T) V so the 1.6 GB
// scores tensor never touches HBM.
// ---------------------------------------------------------------------------

#define USE_ASYNC 1   // global_load_async_to_lds_b128 + s_wait_asynccnt path

typedef __bf16 bf16_t;
typedef __attribute__((ext_vector_type(16))) __bf16 v16bf;
typedef __attribute__((ext_vector_type(8)))  __bf16 v8bf;
typedef __attribute__((ext_vector_type(4)))  __bf16 v4bf;
typedef __attribute__((ext_vector_type(8)))  float  v8f;

#define D_MODEL 2048
#define N_HEADS 16
#define DIM_K   128
#define DIM_V   128
#define D_MID   6144
#define NTOK    4096   // B*T = 2*2048

static __device__ __forceinline__ v16bf frag_cat(v8bf lo, v8bf hi) {
  return __builtin_shufflevector(lo, hi, 0,1,2,3,4,5,6,7,8,9,10,11,12,13,14,15);
}

// A-operand fragment (16x32 bf16), row-major source [row][k].
// ISA layout: lane<16 holds K {k0..k0+7} ∪ {k0+16..k0+23}; lane>=16 is +8.
static __device__ __forceinline__ v16bf load_afrag(const bf16_t* row, int k0, int lh) {
  v8bf lo = *(const v8bf*)(row + k0 + lh * 8);
  v8bf hi = *(const v8bf*)(row + k0 + 16 + lh * 8);
  return frag_cat(lo, hi);
}

// B-operand fragment (32x16 bf16) from N-major source [n][k].
// ISA layout: lane<16 holds K {k0..k0+7} ∪ {k0+16..k0+23}; lane>=16 is +8.
static __device__ __forceinline__ v16bf load_bfrag(const bf16_t* row, int k0, int lh) {
  v8bf lo = *(const v8bf*)(row + k0 + lh * 8);
  v8bf hi = *(const v8bf*)(row + k0 + 16 + lh * 8);
  return frag_cat(lo, hi);
}

static __device__ __forceinline__ v8f wmma_bf16(v16bf a, v16bf b, v8f c) {
  return __builtin_amdgcn_wmma_f32_16x16x32_bf16(false, a, false, b, (short)0, c,
                                                 false, false);
}

// silu with fast reciprocal (v_rcp_f32) instead of IEEE div chains
static __device__ __forceinline__ float fast_silu(float x) {
  return x * __builtin_amdgcn_rcpf(1.0f + __expf(-x));
}

#if USE_ASYNC
// ISA §10.7/§15.18: the LDS address is taken per lane from the VDST VGPR (plus offset);
// low 32 bits of a generic LDS flat pointer are the LDS byte offset (§10.2).
static __device__ __forceinline__ void async_b128(const bf16_t* g, const bf16_t* l) {
  unsigned loff = (unsigned)(unsigned long long)l;
  asm volatile("global_load_async_to_lds_b128 %0, %1, off"
               :: "v"(loff), "v"(g) : "memory");
}
#define WAIT_ASYNC(n) asm volatile("s_wait_asynccnt %0" :: "n"(n) : "memory")
#endif

// ---------------------------------------------------------------------------
// fp32 -> bf16 straight conversion (vectorized: float4 in, 4x bf16 out)
// ---------------------------------------------------------------------------
__global__ void cvt_f32_to_bf16(const float* __restrict__ s, bf16_t* __restrict__ d,
                                int n4) {
  int i = blockIdx.x * blockDim.x + threadIdx.x;
  if (i < n4) {
    float4 f = ((const float4*)s)[i];
    v4bf o;
    o[0] = (bf16_t)f.x; o[1] = (bf16_t)f.y; o[2] = (bf16_t)f.z; o[3] = (bf16_t)f.w;
    ((v4bf*)d)[i] = o;
  }
}

// ---------------------------------------------------------------------------
// fp32 (RxC row-major) -> bf16 (CxR row-major) tiled transpose, batched by z.
// ---------------------------------------------------------------------------
__global__ void transpose_cvt(const float* __restrict__ src, bf16_t* __restrict__ dst,
                              int R, int C) {
  __shared__ float tile[32][33];
  const float* s = src + (size_t)blockIdx.z * R * C;
  bf16_t*      d = dst + (size_t)blockIdx.z * R * C;
  int c0 = blockIdx.x * 32, r0 = blockIdx.y * 32;
#pragma unroll
  for (int i = 0; i < 4; ++i) {
    int r = r0 + threadIdx.y + i * 8;
    tile[threadIdx.y + i * 8][threadIdx.x] = s[(size_t)r * C + c0 + threadIdx.x];
  }
  __syncthreads();
#pragma unroll
  for (int i = 0; i < 4; ++i) {
    int c = c0 + threadIdx.y + i * 8;
    d[(size_t)c * R + r0 + threadIdx.x] = (bf16_t)tile[threadIdx.x][threadIdx.y + i * 8];
  }
}

// ---------------------------------------------------------------------------
// Tiled WMMA GEMM: C[M,N] = A[M,K] * B[N,K]^T  (B stored N-major).
// 128x128 block tile, 8 waves, each wave a 32x64 tile (2x4 accums).
// Double-buffered async-to-LDS staging of 64-deep K slabs.
// ---------------------------------------------------------------------------
template <bool OUT_BF16>
__global__ __launch_bounds__(256) void gemm_wmma(const bf16_t* __restrict__ A,
                                                 const bf16_t* __restrict__ Bm,
                                                 void* __restrict__ Cv,
                                                 int M, int N, int Kd) {
  __shared__ __align__(16) bf16_t lA[2][128][64];
  __shared__ __align__(16) bf16_t lB[2][128][64];
  const int tid  = threadIdx.x;
  const int lane = tid & 31, wave = tid >> 5;
  const int r = lane & 15, lh = lane >> 4;
  const int row0 = blockIdx.y * 128, col0 = blockIdx.x * 128;
  const int mw = (wave & 3) * 32;
  const int nw = (wave >> 2) * 64;

  v8f acc[2][4];
#pragma unroll
  for (int mi = 0; mi < 2; ++mi)
#pragma unroll
    for (int ni = 0; ni < 4; ++ni) acc[mi][ni] = v8f{};

  const int lrow = tid >> 1;
  const int lko  = (tid & 1) * 32;
  const bf16_t* gA = A  + (size_t)(row0 + lrow) * Kd + lko;
  const bf16_t* gB = Bm + (size_t)(col0 + lrow) * Kd + lko;

  auto stage = [&](int buf, int k0) {
#if USE_ASYNC
#pragma unroll
    for (int q = 0; q < 4; ++q) {
      async_b128(gA + k0 + q * 8, &lA[buf][lrow][lko + q * 8]);
      async_b128(gB + k0 + q * 8, &lB[buf][lrow][lko + q * 8]);
    }
#else
    const v8bf* sa = (const v8bf*)(gA + k0);
    v8bf* da = (v8bf*)&lA[buf][lrow][lko];
    da[0] = sa[0]; da[1] = sa[1]; da[2] = sa[2]; da[3] = sa[3];
    const v8bf* sb = (const v8bf*)(gB + k0);
    v8bf* db = (v8bf*)&lB[buf][lrow][lko];
    db[0] = sb[0]; db[1] = sb[1]; db[2] = sb[2]; db[3] = sb[3];
#endif
  };

  const int nk = Kd >> 6;  // 64-deep slabs
  stage(0, 0);
  for (int i = 0; i < nk; ++i) {
    const int cur  = i & 1;
    const bool more = (i + 1 < nk);
    if (more) stage(cur ^ 1, (i + 1) * 64);
#if USE_ASYNC
    // next slab's 8 b128s may be outstanding; in-order completion => <=8
    // guarantees this wave's current slab landed in LDS.
    if (more) WAIT_ASYNC(8); else WAIT_ASYNC(0);
#endif
    __syncthreads();
#pragma unroll
    for (int kk = 0; kk < 64; kk += 32) {
      v16bf af[2], bfv[4];
#pragma unroll
      for (int mi = 0; mi < 2; ++mi)
        af[mi] = load_afrag(&lA[cur][mw + mi * 16 + r][0], kk, lh);
#pragma unroll
      for (int ni = 0; ni < 4; ++ni)
        bfv[ni] = load_bfrag(&lB[cur][nw + ni * 16 + r][0], kk, lh);
#pragma unroll
      for (int mi = 0; mi < 2; ++mi)
#pragma unroll
        for (int ni = 0; ni < 4; ++ni)
          acc[mi][ni] = wmma_bf16(af[mi], bfv[ni], acc[mi][ni]);
    }
    __syncthreads();  // all reads of buf `cur` done before it is re-staged
  }

  // C layout: VGPR j -> m_local = j + 8*lh, n_local = r
#pragma unroll
  for (int mi = 0; mi < 2; ++mi)
#pragma unroll
    for (int ni = 0; ni < 4; ++ni)
#pragma unroll
      for (int j = 0; j < 8; ++j) {
        int gm = row0 + mw + mi * 16 + j + 8 * lh;
        int gn = col0 + nw + ni * 16 + r;
        float v = acc[mi][ni][j];
        if (OUT_BF16) ((bf16_t*)Cv)[(size_t)gm * N + gn] = (bf16_t)v;
        else          ((float*)Cv)[(size_t)gm * N + gn]  = v;
      }
}

// ---------------------------------------------------------------------------
// Fused per-head: mid[tok, h*128+v] = silu(Q[tok,h,:]·K[h]^T) · V[h]
// Block: (128 tokens, 1 head), 8 waves x 16 tokens. Streams D_MID in 64-row
// double-buffered chunks; Q fragments and output accumulators in registers.
// ---------------------------------------------------------------------------
__global__ __launch_bounds__(256) void fused_mhf(const bf16_t* __restrict__ Q,
                                                 const bf16_t* __restrict__ Kw,
                                                 const bf16_t* __restrict__ Vt,
                                                 bf16_t* __restrict__ mid) {
  __shared__ __align__(16) bf16_t lK[2][64][128];   // K chunk  [m][k]
  __shared__ __align__(16) bf16_t lV[2][128][64];   // V^T chunk [v][m]
  __shared__ __align__(16) bf16_t lS[8][16][64];    // per-wave silu(S) staging

  const int tid  = threadIdx.x, lane = tid & 31, wave = tid >> 5;
  const int r = lane & 15, lh = lane >> 4;
  const int h = blockIdx.y;
  const int gtok = blockIdx.x * 128 + wave * 16;

  // Resident Q fragments (A operand), 16 tokens x 128 k
  v16bf qf[4];
  {
    const bf16_t* qrow = Q + (size_t)(gtok + r) * (N_HEADS * DIM_K) + h * DIM_K;
#pragma unroll
    for (int i = 0; i < 4; ++i) qf[i] = load_afrag(qrow, i * 32, lh);
  }

  v8f oacc[8];
#pragma unroll
  for (int i = 0; i < 8; ++i) oacc[i] = v8f{};

  const bf16_t* Kh = Kw + (size_t)h * D_MID * DIM_K;
  const bf16_t* Vh = Vt + (size_t)h * DIM_V * D_MID;
  const int krow = tid >> 2, kko = (tid & 3) * 32;  // lK coop coords (64B/thr)
  const int vrow = tid >> 1, vmo = (tid & 1) * 32;  // lV coop coords (64B/thr)

  auto stage = [&](int buf, int mc) {
#if USE_ASYNC
#pragma unroll
    for (int q = 0; q < 4; ++q) {
      async_b128(Kh + (size_t)(mc + krow) * DIM_K + kko + q * 8,
                 &lK[buf][krow][kko + q * 8]);
      async_b128(Vh + (size_t)vrow * D_MID + mc + vmo + q * 8,
                 &lV[buf][vrow][vmo + q * 8]);
    }
#else
    const v8bf* sk = (const v8bf*)(Kh + (size_t)(mc + krow) * DIM_K + kko);
    v8bf* dk = (v8bf*)&lK[buf][krow][kko];
    dk[0] = sk[0]; dk[1] = sk[1]; dk[2] = sk[2]; dk[3] = sk[3];
    const v8bf* sv = (const v8bf*)(Vh + (size_t)vrow * D_MID + mc + vmo);
    v8bf* dv = (v8bf*)&lV[buf][vrow][vmo];
    dv[0] = sv[0]; dv[1] = sv[1]; dv[2] = sv[2]; dv[3] = sv[3];
#endif
  };

  stage(0, 0);
  const int nc = D_MID / 64;
  for (int i = 0; i < nc; ++i) {
    const int cur = i & 1;
    const bool more = (i + 1 < nc);
    if (more) stage(cur ^ 1, (i + 1) * 64);
#if USE_ASYNC
    if (more) WAIT_ASYNC(8); else WAIT_ASYNC(0);
#endif
    __syncthreads();

    // GEMM1: S(16x64) = Q(16x128) @ Kchunk^T, fused silu, C-layout -> LDS
#pragma unroll
    for (int nt = 0; nt < 4; ++nt) {
      v8f s = v8f{};
#pragma unroll
      for (int ks = 0; ks < 4; ++ks)
        s = wmma_bf16(qf[ks], load_bfrag(&lK[cur][nt * 16 + r][0], ks * 32, lh), s);
#pragma unroll
      for (int j = 0; j < 8; ++j)
        lS[wave][j + 8 * lh][nt * 16 + r] = (bf16_t)fast_silu(s[j]);
    }
    // wave-local LDS store->load: DS ops in-order per wave, compiler inserts
    // the DScnt wait; no cross-wave barrier needed for lS.

    // GEMM2: oacc(16x128) += silu(S)(16x64) @ Vchunk(64x128)
#pragma unroll
    for (int kt = 0; kt < 2; ++kt) {
      v16bf sf = load_afrag(&lS[wave][r][0], kt * 32, lh);
#pragma unroll
      for (int vt = 0; vt < 8; ++vt)
        oacc[vt] = wmma_bf16(sf, load_bfrag(&lV[cur][vt * 16 + r][0], kt * 32, lh),
                             oacc[vt]);
    }
    __syncthreads();  // protect buf `cur` before it is re-staged
  }

#pragma unroll
  for (int vt = 0; vt < 8; ++vt)
#pragma unroll
    for (int j = 0; j < 8; ++j) {
      int gm = gtok + j + 8 * lh;
      int gn = h * DIM_V + vt * 16 + r;
      mid[(size_t)gm * (N_HEADS * DIM_V) + gn] = (bf16_t)oacc[vt][j];
    }
}

// ---------------------------------------------------------------------------
extern "C" void kernel_launch(void* const* d_in, const int* in_sizes, int n_in,
                              void* d_out, int out_size, void* d_ws, size_t ws_size,
                              hipStream_t stream) {
  (void)in_sizes; (void)n_in; (void)out_size;
  const float* x  = (const float*)d_in[0];
  const float* Wq = (const float*)d_in[1];
  const float* Wo = (const float*)d_in[2];
  const float* K  = (const float*)d_in[3];
  const float* V  = (const float*)d_in[4];

  char* ws = (char*)d_ws;
  size_t off = 0;
  auto alloc = [&](size_t bytes) -> void* {
    void* p = ws + off;
    off += (bytes + 255) & ~(size_t)255;
    return p;
  };
  bf16_t* xb   = (bf16_t*)alloc((size_t)NTOK * D_MODEL * 2);            // x  bf16
  bf16_t* wqt  = (bf16_t*)alloc((size_t)D_MODEL * D_MODEL * 2);         // Wq^T
  bf16_t* wot  = (bf16_t*)alloc((size_t)D_MODEL * D_MODEL * 2);         // Wo^T
  bf16_t* kb   = (bf16_t*)alloc((size_t)N_HEADS * D_MID * DIM_K * 2);   // K bf16
  bf16_t* vtb  = (bf16_t*)alloc((size_t)N_HEADS * DIM_V * D_MID * 2);   // V^T bf16
  bf16_t* qb   = (bf16_t*)alloc((size_t)NTOK * N_HEADS * DIM_K * 2);    // Q bf16
  bf16_t* midb = (bf16_t*)alloc((size_t)NTOK * N_HEADS * DIM_V * 2);    // S@V bf16
  if (off > ws_size) return;  // workspace too small (~118 MB needed)

  // Stage 0: precision conversion / layout transforms
  int n4x = NTOK * D_MODEL / 4;
  cvt_f32_to_bf16<<<(n4x + 255) / 256, 256, 0, stream>>>(x, xb, n4x);
  int n4k = N_HEADS * D_MID * DIM_K / 4;
  cvt_f32_to_bf16<<<(n4k + 255) / 256, 256, 0, stream>>>(K, kb, n4k);
  transpose_cvt<<<dim3(D_MODEL / 32, D_MODEL / 32, 1), dim3(32, 8), 0, stream>>>(
      Wq, wqt, D_MODEL, D_MODEL);
  transpose_cvt<<<dim3(D_MODEL / 32, D_MODEL / 32, 1), dim3(32, 8), 0, stream>>>(
      Wo, wot, N_HEADS * DIM_V, D_MODEL);
  transpose_cvt<<<dim3(DIM_V / 32, D_MID / 32, N_HEADS), dim3(32, 8), 0, stream>>>(
      V, vtb, D_MID, DIM_V);

  // Stage 1: Q = x @ Wq  (bf16 out)
  gemm_wmma<true><<<dim3(D_MODEL / 128, NTOK / 128), 256, 0, stream>>>(
      xb, wqt, qb, NTOK, D_MODEL, D_MODEL);

  // Stage 2+3: fused per-head silu(Q K^T) V  (scores never hit HBM)
  fused_mhf<<<dim3(NTOK / 128, N_HEADS), 256, 0, stream>>>(qb, kb, vtb, midb);

  // Stage 4: out = mid @ Wo  (f32 out)
  gemm_wmma<false><<<dim3(D_MODEL / 128, NTOK / 128), 256, 0, stream>>>(
      midb, wot, d_out, NTOK, D_MODEL, D_MODEL);
}
